// DimeNet_58196806860852
// MI455X (gfx1250) — matrix operand order repaired and checked by reading, not compiled, not run on hardware
//
#include <hip/hip_runtime.h>

typedef __attribute__((ext_vector_type(16))) __bf16 v16bf;
typedef __attribute__((ext_vector_type(2)))  __bf16 v2bf;
typedef __attribute__((ext_vector_type(8)))  float  v8f;

#define HDIM 256
#define RDIM 6
#define SRDIM 42
#define BILD 8
#define NBLK 3
#define OUTD 12
#define NOLD 3
#define NNODE 8000
#define NEDGE 80000
#define NTRIP 100000

#define LDS_A 36   // padded row stride (elements) for 128x32 A tile
#define LDS_W 36   // padded row stride (elements) for 64x32 W tile (stored [n][k])

// ---------- helpers ----------
// native fptrunc -> bf16 (RTNE on gfx1250)
__device__ __forceinline__ unsigned short f2bf(float f) {
  __bf16 h = (__bf16)f;
  return __builtin_bit_cast(unsigned short, h);
}
// pack two floats into one dword of bf16 pair -> single v_cvt_pk_bf16_f32
__device__ __forceinline__ unsigned pack2bf(float a, float b) {
  v2bf v;
  v[0] = (__bf16)a;
  v[1] = (__bf16)b;
  return __builtin_bit_cast(unsigned, v);
}
__device__ __forceinline__ __bf16 us2bf(unsigned short s) {
  return __builtin_bit_cast(__bf16, s);
}
__device__ __forceinline__ float swishf(float v) {
  return v / (1.0f + __expf(-v));
}

// A fragment: 16x32 bf16, lane m = lane&15, khalf = (lane>>4)*8
__device__ __forceinline__ v16bf load_fragA(const unsigned short* sA, int arow, int khA) {
  v16bf a;
#pragma unroll
  for (int v = 0; v < 8; ++v) {
    int klo = (v < 4 ? 2 * v : 16 + 2 * (v - 4)) + khA;
    unsigned u = *(const unsigned*)(sA + arow * LDS_A + klo);
    a[2 * v]     = us2bf((unsigned short)(u & 0xFFFFu));
    a[2 * v + 1] = us2bf((unsigned short)(u >> 16));
  }
  return a;
}
// B fragment: 32x16 bf16, lane n = lane&15, khalf = (lane>>4)*8; sW stored [n][k]
__device__ __forceinline__ v16bf load_fragB(const unsigned short* sW, int nrow, int khB) {
  v16bf b;
#pragma unroll
  for (int v = 0; v < 8; ++v) {
    int k2 = (v < 4 ? 2 * v : 16 + 2 * (v - 4)) + khB;
    unsigned u = *(const unsigned*)(sW + nrow * LDS_W + k2);
    b[2 * v]     = us2bf((unsigned short)(u & 0xFFFFu));
    b[2 * v + 1] = us2bf((unsigned short)(u >> 16));
  }
  return b;
}

// stage 128x32 f32->bf16 A tile: 1024 float4 across 256 threads (4 iters, b128 loads)
__device__ __forceinline__ void stage_A(const float* __restrict__ A, unsigned short* sA,
                                        int blockM, int kk, int M, int K, int tid) {
#pragma unroll
  for (int it = 0; it < 4; ++it) {
    int q = tid + it * 256;
    int r = q >> 3, c4 = (q & 7) << 2;
    int row = blockM + r;
    float4 v = make_float4(0.f, 0.f, 0.f, 0.f);
    if (row < M) v = *(const float4*)(A + (size_t)row * K + kk + c4);
    unsigned* dst = (unsigned*)(sA + r * LDS_A + c4);   // 8B aligned (36*r + c4)
    dst[0] = pack2bf(v.x, v.y);
    dst[1] = pack2bf(v.z, v.w);
  }
}
// stage 32x64 W tile transposed into sW[n][k]: 512 float4 across 256 threads (2 iters)
__device__ __forceinline__ void stage_W(const float* __restrict__ W, unsigned short* sW,
                                        int blockN, int kk, int N, int tid) {
#pragma unroll
  for (int it = 0; it < 2; ++it) {
    int q = tid + it * 256;
    int kr = q >> 4, n4 = (q & 15) << 2;
    float4 v = *(const float4*)(W + (size_t)(kk + kr) * N + blockN + n4);
    sW[(n4 + 0) * LDS_W + kr] = f2bf(v.x);
    sW[(n4 + 1) * LDS_W + kr] = f2bf(v.y);
    sW[(n4 + 2) * LDS_W + kr] = f2bf(v.z);
    sW[(n4 + 3) * LDS_W + kr] = f2bf(v.w);
  }
}

// ---------- elementwise / small-K kernels ----------
__global__ void k_rbf(const float* __restrict__ dist, const float* __restrict__ freq,
                      float* __restrict__ rbf, int E) {
  int e = blockIdx.x * blockDim.x + threadIdx.x;
  if (e >= E) return;
  float d = dist[e] * (1.0f / 5.0f);
  float d2 = d * d;
  float x5 = d2 * d2 * d;                   // d^5   (p=6)
  float env = 1.0f / d + (-28.0f) * x5 + 48.0f * x5 * d + (-21.0f) * x5 * d2;
#pragma unroll
  for (int r = 0; r < RDIM; ++r) rbf[e * RDIM + r] = env * __sinf(freq[r] * d);
}

// out[m, outOff+n (ld outLD)] = [swish]( A[MxK] @ W[KxN] [+bias] ) [* mul[MxN]]
__global__ void k_smallk(const float* __restrict__ A, const float* __restrict__ W,
                         const float* __restrict__ bias, const float* __restrict__ mul,
                         float* __restrict__ out, int M, int N, int K,
                         int outLD, int outOff, int actSwish) {
  int t = blockIdx.x * blockDim.x + threadIdx.x;
  if (t >= M * N) return;
  int m = t / N, n = t % N;
  float acc = 0.0f;
  for (int k = 0; k < K; ++k) acc += A[m * K + k] * W[k * N + n];
  if (bias) acc += bias[n];
  if (actSwish) acc = swishf(acc);
  if (mul) acc *= mul[m * N + n];
  out[m * outLD + outOff + n] = acc;
}

__global__ void k_embed_cat(const int* __restrict__ z, const int* __restrict__ ii,
                            const int* __restrict__ jj, const float* __restrict__ table,
                            float* __restrict__ acat) {
  int t = blockIdx.x * blockDim.x + threadIdx.x;          // E * H/4 threads
  if (t >= NEDGE * (HDIM / 4)) return;
  int e = t / (HDIM / 4), h4 = (t % (HDIM / 4)) * 4;
  int zi = z[ii[e]], zj = z[jj[e]];
  float4 vi = *(const float4*)(table + zi * HDIM + h4);
  float4 vj = *(const float4*)(table + zj * HDIM + h4);
  *(float4*)(acat + (size_t)e * (3 * HDIM) + h4)        = vi;
  *(float4*)(acat + (size_t)e * (3 * HDIM) + HDIM + h4) = vj;
}

__global__ void k_gather(const float* __restrict__ src, const int* __restrict__ idx,
                         float* __restrict__ dst, int rows) {
  int t = blockIdx.x * blockDim.x + threadIdx.x;          // rows * H/4 threads
  if (t >= rows * (HDIM / 4)) return;
  int r = t / (HDIM / 4), h4 = (t % (HDIM / 4)) * 4;
  *(float4*)(dst + (size_t)r * HDIM + h4) =
      *(const float4*)(src + (size_t)idx[r] * HDIM + h4);
}

__global__ void k_segsum(const float* __restrict__ src, const int* __restrict__ seg,
                         float* __restrict__ dst, int rows) {
  int t = blockIdx.x * blockDim.x + threadIdx.x;
  if (t >= rows * HDIM) return;
  int r = t / HDIM, h = t % HDIM;
  atomicAdd(dst + (size_t)seg[r] * HDIM + h, src[t]);
}

__global__ void k_fill(float* __restrict__ p, int n4) {   // n4 = count/4
  int t = blockIdx.x * blockDim.x + threadIdx.x;
  if (t < n4) *(float4*)(p + 4 * (size_t)t) = make_float4(0.f, 0.f, 0.f, 0.f);
}

// BILW[j][l][i] = bilin[i][j][l]   (bilin is (H,BIL,H))
__global__ void k_bilin_tr(const float* __restrict__ bl, float* __restrict__ out) {
  int t = blockIdx.x * blockDim.x + threadIdx.x;
  if (t >= BILD * HDIM * HDIM) return;
  int j = t / (HDIM * HDIM);
  int rem = t % (HDIM * HDIM);
  int l = rem / HDIM, i2 = rem % HDIM;
  out[t] = bl[(i2 * BILD + j) * HDIM + l];
}

__global__ void k_out_small(const float* __restrict__ node, const float* __restrict__ w,
                            float* __restrict__ P, int Nn, int accum) {
  int t = blockIdx.x * blockDim.x + threadIdx.x;
  if (t >= Nn * OUTD) return;
  int n = t / OUTD, o = t % OUTD;
  float s = 0.0f;
  for (int h = 0; h < HDIM; ++h) s += node[n * HDIM + h] * w[h * OUTD + o];
  P[t] = accum ? (P[t] + s) : s;
}

// ---------- WMMA GEMM: out = epi(A[MxK] @ W[KxN]) ----------
// block tile 128x64, 8 wave32 waves each own 16 rows x 64 cols (4 wmma / K-step).
// epi: (+bias) -> (swish) -> (*mul) -> (+add).  K%32==0, N%64==0.
__global__ __launch_bounds__(256)
void k_gemm(const float* __restrict__ A, const float* __restrict__ W,
            const float* __restrict__ bias, const float* __restrict__ mul,
            const float* __restrict__ add, float* __restrict__ out,
            int M, int K, int N, int actSwish) {
  __shared__ unsigned short sA[128 * LDS_A];
  __shared__ unsigned short sW[64 * LDS_W];
  const int tid = threadIdx.x;
  const int wave = tid >> 5, lane = tid & 31;
  const int blockM = blockIdx.x * 128, blockN = blockIdx.y * 64;
  const int khA = (lane >> 4) * 8, khB = (lane >> 4) * 8;
  const int mloc = lane & 15, nB = lane & 15;
  v8f c[4] = {{}, {}, {}, {}};
  for (int kk = 0; kk < K; kk += 32) {
    __syncthreads();
    stage_A(A, sA, blockM, kk, M, K, tid);
    stage_W(W, sW, blockN, kk, N, tid);
    __syncthreads();
    v16bf a = load_fragA(sA, wave * 16 + mloc, khA);
#pragma unroll
    for (int t = 0; t < 4; ++t) {
      v16bf b = load_fragB(sW, t * 16 + nB, khB);
      c[t] = __builtin_amdgcn_wmma_f32_16x16x32_bf16(false, a, false, b, (short)0,
                                                     c[t], false, false);
    }
  }
  const int mbase = (lane >> 4) * 8;
#pragma unroll
  for (int r = 0; r < 8; ++r) {
    int row = blockM + wave * 16 + r + mbase;
    if (row < M) {
#pragma unroll
      for (int t = 0; t < 4; ++t) {
        int col = blockN + t * 16 + nB;
        float v = c[t][r];
        if (bias) v += bias[col];
        if (actSwish) v = swishf(v);
        if (mul) v *= mul[(size_t)row * N + col];
        if (add) v += add[(size_t)row * N + col];
        out[(size_t)row * N + col] = v;
      }
    }
  }
}

// ---------- WMMA bilinear: msg[w,i] = sum_j sbfb[w,j] * (G[w,:] @ BW[j][:,i]) ----------
__global__ __launch_bounds__(256)
void k_bilin(const float* __restrict__ G, const float* __restrict__ BW,
             const float* __restrict__ sbfb, float* __restrict__ out, int M) {
  __shared__ unsigned short sA[128 * LDS_A];
  __shared__ unsigned short sW[64 * LDS_W];
  const int tid = threadIdx.x;
  const int wave = tid >> 5, lane = tid & 31;
  const int blockM = blockIdx.x * 128, blockN = blockIdx.y * 64;
  const int khA = (lane >> 4) * 8, khB = (lane >> 4) * 8;
  const int mloc = lane & 15, nB = lane & 15;
  const int mbase = (lane >> 4) * 8;
  float fin[4][8] = {};
  for (int j = 0; j < BILD; ++j) {
    const float* W = BW + j * HDIM * HDIM;
    v8f c[4] = {{}, {}, {}, {}};
    for (int kk = 0; kk < HDIM; kk += 32) {
      __syncthreads();
      stage_A(G, sA, blockM, kk, M, HDIM, tid);
      stage_W(W, sW, blockN, kk, HDIM, tid);
      __syncthreads();
      v16bf a = load_fragA(sA, wave * 16 + mloc, khA);
#pragma unroll
      for (int t = 0; t < 4; ++t) {
        v16bf b = load_fragB(sW, t * 16 + nB, khB);
        c[t] = __builtin_amdgcn_wmma_f32_16x16x32_bf16(false, a, false, b, (short)0,
                                                       c[t], false, false);
      }
    }
#pragma unroll
    for (int r = 0; r < 8; ++r) {
      int row = blockM + wave * 16 + r + mbase;
      float s = (row < M) ? sbfb[row * BILD + j] : 0.0f;
#pragma unroll
      for (int t = 0; t < 4; ++t) fin[t][r] += s * c[t][r];
    }
  }
#pragma unroll
  for (int r = 0; r < 8; ++r) {
    int row = blockM + wave * 16 + r + mbase;
    if (row < M) {
#pragma unroll
      for (int t = 0; t < 4; ++t)
        out[(size_t)row * HDIM + blockN + t * 16 + nB] = fin[t][r];
    }
  }
}

// ---------- driver ----------
static inline int cdiv(long a, long b) { return (int)((a + b - 1) / b); }

extern "C" void kernel_launch(void* const* d_in, const int* in_sizes, int n_in,
                              void* d_out, int out_size, void* d_ws, size_t ws_size,
                              hipStream_t stream) {
  (void)in_sizes; (void)n_in; (void)out_size; (void)ws_size;
  const int*   z         = (const int*)d_in[0];
  const float* dist      = (const float*)d_in[1];
  const float* sbf       = (const float*)d_in[2];
  const int*   iIdx      = (const int*)d_in[3];
  const int*   jIdx      = (const int*)d_in[4];
  const int*   idx_kj    = (const int*)d_in[5];
  const int*   idx_ji    = (const int*)d_in[6];
  const float* freq      = (const float*)d_in[7];
  const float* emb_table = (const float*)d_in[8];
  const float* emb_rbf_w = (const float*)d_in[9];
  const float* emb_rbf_b = (const float*)d_in[10];
  const float* emb_lin_w = (const float*)d_in[11];
  const float* emb_lin_b = (const float*)d_in[12];
  const float* ib_rbf_w  = (const float*)d_in[13];
  const float* ib_sbf_w  = (const float*)d_in[14];
  const float* ib_kj_w   = (const float*)d_in[15];
  const float* ib_kj_b   = (const float*)d_in[16];
  const float* ib_ji_w   = (const float*)d_in[17];
  const float* ib_ji_b   = (const float*)d_in[18];
  const float* ib_bilin  = (const float*)d_in[19];
  const float* ib_bs_w1  = (const float*)d_in[20];
  const float* ib_bs_b1  = (const float*)d_in[21];
  const float* ib_bs_w2  = (const float*)d_in[22];
  const float* ib_bs_b2  = (const float*)d_in[23];
  const float* ib_lin_w  = (const float*)d_in[24];
  const float* ib_lin_b  = (const float*)d_in[25];
  const float* ib_as_w1  = (const float*)d_in[26];
  const float* ib_as_b1  = (const float*)d_in[27];
  const float* ib_as_w2  = (const float*)d_in[28];
  const float* ib_as_b2  = (const float*)d_in[29];
  const float* ob_rbf_w  = (const float*)d_in[30];
  const float* ob_lin_w  = (const float*)d_in[31];
  const float* ob_lin_b  = (const float*)d_in[32];
  const float* ob_out_w  = (const float*)d_in[33];
  float* P = (float*)d_out;

  const size_t EH = (size_t)NEDGE * HDIM;
  const size_t TH = (size_t)NTRIP * HDIM;
  float* ws   = (float*)d_ws;
  float* RBF  = ws;                                    // E*6
  float* XA   = RBF + (size_t)NEDGE * RDIM;            // E*H
  float* XB   = XA + EH;                               // E*H
  float* XJ   = XB + EH;                               // E*H
  float* XK   = XJ + EH;                               // E*H (x_kj, then hh)
  float* RB   = XK + EH;                               // E*H
  float* TMP  = RB + EH;                               // E*H
  float* SBFB = TMP + EH;                              // T*8
  float* NODE = SBFB + (size_t)NTRIP * BILD;           // N*H
  float* NODE2= NODE + (size_t)NNODE * HDIM;           // N*H
  float* BILW = NODE2 + (size_t)NNODE * HDIM;          // 8*H*H
  float* ACAT = BILW + (size_t)BILD * HDIM * HDIM;     // max(E*3H, 2*T*H)
  float* Gbuf = ACAT;                                  // T*H (after embedding)
  float* MSG  = ACAT + TH;                             // T*H

  const dim3 B(256);
  const dim3 gEH(cdiv((long)NEDGE * HDIM, 256));
  const dim3 gTH(cdiv((long)NTRIP * HDIM, 256));
  const dim3 gEH4(cdiv((long)NEDGE * (HDIM / 4), 256));
  const dim3 gTH4(cdiv((long)NTRIP * (HDIM / 4), 256));
  const dim3 gGemmE(cdiv(NEDGE, 128), HDIM / 64);
  const dim3 gGemmT(cdiv(NTRIP, 128), HDIM / 64);
  const dim3 gGemmN(cdiv(NNODE, 128), HDIM / 64);

  // 1) Bessel RBF
  k_rbf<<<cdiv(NEDGE, 256), B, 0, stream>>>(dist, freq, RBF, NEDGE);
  // 2) Embedding: rbf_e -> ACAT[:,512:768], h[i]/h[j] -> ACAT[:,0:512]
  k_smallk<<<gEH, B, 0, stream>>>(RBF, emb_rbf_w, emb_rbf_b, nullptr, ACAT,
                                  NEDGE, HDIM, RDIM, 3 * HDIM, 2 * HDIM, 1);
  k_embed_cat<<<gEH4, B, 0, stream>>>(z, iIdx, jIdx, emb_table, ACAT);
  k_gemm<<<gGemmE, B, 0, stream>>>(ACAT, emb_lin_w, emb_lin_b, nullptr, nullptr,
                                   XA, NEDGE, 3 * HDIM, HDIM, 1);

  float* xcur = XA;
  float* xnext = XB;

  auto output_block = [&](int bb, const float* xe, int accum) {
    k_smallk<<<gEH, B, 0, stream>>>(RBF, ob_rbf_w + (size_t)bb * RDIM * HDIM,
                                    nullptr, xe, TMP, NEDGE, HDIM, RDIM, HDIM, 0, 0);
    k_fill<<<cdiv((long)NNODE * HDIM / 4, 256), B, 0, stream>>>(NODE, NNODE * HDIM / 4);
    k_segsum<<<gEH, B, 0, stream>>>(TMP, iIdx, NODE, NEDGE);
    const float* lw = ob_lin_w + (size_t)bb * NOLD * HDIM * HDIM;
    const float* lb = ob_lin_b + (size_t)bb * NOLD * HDIM;
    k_gemm<<<gGemmN, B, 0, stream>>>(NODE, lw, lb, nullptr, nullptr, NODE2,
                                     NNODE, HDIM, HDIM, 1);
    k_gemm<<<gGemmN, B, 0, stream>>>(NODE2, lw + HDIM * HDIM, lb + HDIM, nullptr,
                                     nullptr, NODE, NNODE, HDIM, HDIM, 1);
    k_gemm<<<gGemmN, B, 0, stream>>>(NODE, lw + 2 * HDIM * HDIM, lb + 2 * HDIM,
                                     nullptr, nullptr, NODE2, NNODE, HDIM, HDIM, 1);
    k_out_small<<<cdiv((long)NNODE * OUTD, 256), B, 0, stream>>>(
        NODE2, ob_out_w + (size_t)bb * HDIM * OUTD, P, NNODE, accum);
  };

  output_block(0, XA, 0);

  for (int b = 0; b < NBLK; ++b) {
    // rbf_b [E,H], sbf_b [T,8]
    k_smallk<<<gEH, B, 0, stream>>>(RBF, ib_rbf_w + (size_t)b * RDIM * HDIM,
                                    nullptr, nullptr, RB, NEDGE, HDIM, RDIM, HDIM, 0, 0);
    k_smallk<<<cdiv((long)NTRIP * BILD, 256), B, 0, stream>>>(
        sbf, ib_sbf_w + (size_t)b * SRDIM * BILD, nullptr, nullptr, SBFB,
        NTRIP, BILD, SRDIM, BILD, 0, 0);
    // x_ji, x_kj
    k_gemm<<<gGemmE, B, 0, stream>>>(xcur, ib_ji_w + (size_t)b * HDIM * HDIM,
                                     ib_ji_b + (size_t)b * HDIM, nullptr, nullptr,
                                     XJ, NEDGE, HDIM, HDIM, 1);
    k_gemm<<<gGemmE, B, 0, stream>>>(xcur, ib_kj_w + (size_t)b * HDIM * HDIM,
                                     ib_kj_b + (size_t)b * HDIM, RB, nullptr,
                                     XK, NEDGE, HDIM, HDIM, 1);
    // triplet gather + bilinear message
    k_gather<<<gTH4, B, 0, stream>>>(XK, idx_kj, Gbuf, NTRIP);
    k_bilin_tr<<<cdiv((long)BILD * HDIM * HDIM, 256), B, 0, stream>>>(
        ib_bilin + (size_t)b * HDIM * BILD * HDIM, BILW);
    k_bilin<<<gGemmT, B, 0, stream>>>(Gbuf, BILW, SBFB, MSG, NTRIP);
    // hh = x_ji + segment_sum(msg, idx_ji)
    hipMemcpyAsync(XK, XJ, EH * sizeof(float), hipMemcpyDeviceToDevice, stream);
    k_segsum<<<gTH, B, 0, stream>>>(MSG, idx_ji, XK, NTRIP);
    // before-skip residual layer (NBS=1)
    k_gemm<<<gGemmE, B, 0, stream>>>(XK, ib_bs_w1 + (size_t)b * HDIM * HDIM,
                                     ib_bs_b1 + (size_t)b * HDIM, nullptr, nullptr,
                                     TMP, NEDGE, HDIM, HDIM, 1);
    k_gemm<<<gGemmE, B, 0, stream>>>(TMP, ib_bs_w2 + (size_t)b * HDIM * HDIM,
                                     ib_bs_b2 + (size_t)b * HDIM, nullptr, XK,
                                     XK, NEDGE, HDIM, HDIM, 1);
    // lin: xnext = swish(hh @ lin + b) + x
    k_gemm<<<gGemmE, B, 0, stream>>>(XK, ib_lin_w + (size_t)b * HDIM * HDIM,
                                     ib_lin_b + (size_t)b * HDIM, nullptr, xcur,
                                     xnext, NEDGE, HDIM, HDIM, 1);
    // after-skip residual layers (NAS=2)
    for (int l = 0; l < 2; ++l) {
      size_t wo = ((size_t)b * 2 + l) * HDIM * HDIM;
      size_t bo = ((size_t)b * 2 + l) * HDIM;
      k_gemm<<<gGemmE, B, 0, stream>>>(xnext, ib_as_w1 + wo, ib_as_b1 + bo,
                                       nullptr, nullptr, TMP, NEDGE, HDIM, HDIM, 1);
      k_gemm<<<gGemmE, B, 0, stream>>>(TMP, ib_as_w2 + wo, ib_as_b2 + bo,
                                       nullptr, xnext, xnext, NEDGE, HDIM, HDIM, 1);
    }
    output_block(b + 1, xnext, 1);
    float* t = xcur; xcur = xnext; xnext = t;
  }
}
